// CRF_33011118637753
// MI455X (gfx1250) — matrix operand order repaired and checked by reading, not compiled, not run on hardware
//
#include <hip/hip_runtime.h>
#include <math.h>

typedef __attribute__((ext_vector_type(16))) _Float16 v16h;
typedef __attribute__((ext_vector_type(8)))  float    v8f;

#define Bb   512
#define Ss   512
#define Tt   128
#define ROWS 16    // batch rows per block == WMMA M
#define SCP  20    // scT row stride (floats): 16B-aligned, bank-conflict-free

#define LOG2E 1.44269504088896340736f
#define LN2   0.69314718055994530942f

// ---------------------------------------------------------------------------
// Kernel 1: numerator path score per batch (gather + block reduction)
// ---------------------------------------------------------------------------
__global__ __launch_bounds__(128) void crf_numerator(
    const float* __restrict__ em, const int* __restrict__ tags,
    const int* __restrict__ mask, const float* __restrict__ startT,
    const float* __restrict__ endT, const float* __restrict__ trans,
    float* __restrict__ scoreWS)
{
    __shared__ float red[128];
    __shared__ int   redi[128];
    const int b   = blockIdx.x;
    const int tid = threadIdx.x;

    float acc = 0.f;
    int   cnt = 0;
    for (int s = tid; s < Ss; s += 128) {
        const int mk = mask[b * Ss + s];
        cnt += mk;
        if (s >= 1 && mk > 0) {
            const int tp = tags[b * Ss + s - 1];
            const int tc = tags[b * Ss + s];
            acc += trans[tp * Tt + tc] + em[((size_t)b * Ss + s) * Tt + tc];
        }
    }
    red[tid] = acc; redi[tid] = cnt;
    __syncthreads();
    for (int off = 64; off >= 1; off >>= 1) {
        if (tid < off) { red[tid] += red[tid + off]; redi[tid] += redi[tid + off]; }
        __syncthreads();
    }
    if (tid == 0) {
        const int t0   = tags[b * Ss];
        const int send = redi[0] - 1;               // seq_end = sum(mask) - 1
        const int tl   = tags[b * Ss + send];
        scoreWS[b] = red[0] + startT[t0] + em[(size_t)b * Ss * Tt + t0] + endT[tl];
    }
}

// ---------------------------------------------------------------------------
// Kernel 2: forward scan (denominator). One block = 16 batch rows, 8 waves.
// Per step: P = exp(sc - rowmax) in f16, D = P(16x128) @ expT(128x128) via
// 4 chained v_wmma_f32_16x16x32_f16 per wave (one 16-col N-tile each),
// sc' = rowmax + ln2*log2(D) + emission.  expT B-fragments live in VGPRs
// for the entire kernel.  Scores are column-major (scT[t][r], stride 20)
// so the write-back is 2x ds_load_b128 + arithmetic mask blend +
// 2x ds_store_b128 -- zero control flow, EXEC stays all-ones at every WMMA.
// ---------------------------------------------------------------------------
__global__ __launch_bounds__(256) void crf_forward(
    const float* __restrict__ em, const int* __restrict__ mask,
    const float* __restrict__ startT, const float* __restrict__ endT,
    const float* __restrict__ trans, float* __restrict__ denomWS)
{
    __shared__ float    scT[Tt][SCP];            // scores, column-major
    __shared__ _Float16 pA[ROWS][Tt];            // exp(sc - rowmax), A source
    __shared__ __align__(16) float mS[ROWS];     // row maxima
    __shared__ __align__(16) int   mkS[ROWS];    // per-row mask this step

    const int b0   = blockIdx.x * ROWS;
    const int tid  = threadIdx.x;
    const int wave = tid >> 5;
    const int lane = tid & 31;
    const int half = lane >> 4;          // lane half selects K sub-pattern
    const int l15  = lane & 15;
    const int n0   = wave * 16;          // this wave's N-tile
    const int rgrp = tid >> 4;           // row (0..15) for max/exp stage
    const int sub  = tid & 15;           // 16 threads cooperate per row

    // ---- Preload constant B fragments: exp(transitions). contiguous K order (same as A):
    // within each K=32 chunk, lanes 0-15 hold K=0..15, lanes 16-31 K=16..31,
    // two K per VGPR; N = n0 + (lane&15).
    v16h bfrag[4];
#pragma unroll
    for (int c = 0; c < 4; ++c) {
        v16h bf;
#pragma unroll
        for (int j = 0; j < 8; ++j) {
            const int k0 = 32 * c + 16 * half + 2 * j;
            bf[2 * j]     = (_Float16)__builtin_amdgcn_exp2f(trans[(k0    ) * Tt + n0 + l15] * LOG2E);
            bf[2 * j + 1] = (_Float16)__builtin_amdgcn_exp2f(trans[(k0 + 1) * Tt + n0 + l15] * LOG2E);
        }
        bfrag[c] = bf;
    }

    // ---- init: scT[t][r] = start_transitions[t] + emissions[b0+r, 0, t]
    for (int i = tid; i < ROWS * Tt; i += 256) {
        const int r = i >> 7, t = i & 127;
        scT[t][r] = startT[t] + em[((size_t)(b0 + r) * Ss) * Tt + t];
    }
    __syncthreads();

    for (int s = 1; s < Ss; ++s) {
        // ---- issue this step's emission loads early (used after WMMA):
        // row = 8*half + v, col = n0 + l15  (lanes 0-15: 64B coalesced/row)
        float ev[8];
#pragma unroll
        for (int v = 0; v < 8; ++v) {
            const int row = half * 8 + v;
            ev[v] = em[((size_t)(b0 + row) * Ss + s) * Tt + n0 + l15];
        }
        if (tid < ROWS) mkS[tid] = mask[(b0 + tid) * Ss + s];

        // ---- prefetch next step's emission tile into near cache
        if (s + 1 < Ss) {
            __builtin_prefetch(&em[((size_t)(b0 + rgrp) * Ss + (s + 1)) * Tt + sub * 8], 0, 3);
        }

        // ---- row max (16-lane butterfly), then P = exp2((sc-max)*log2e);
        // score column is read from LDS once and kept in registers.
        float sv[8];
        float mx = -3.402823e38f;
#pragma unroll
        for (int k = 0; k < 8; ++k) {
            sv[k] = scT[sub + 16 * k][rgrp];
            mx = fmaxf(mx, sv[k]);
        }
#pragma unroll
        for (int off = 8; off >= 1; off >>= 1) mx = fmaxf(mx, __shfl_xor(mx, off, 16));
#pragma unroll
        for (int k = 0; k < 8; ++k) {
            pA[rgrp][sub + 16 * k] = (_Float16)__builtin_amdgcn_exp2f((sv[k] - mx) * LOG2E);
        }
        if (sub == 0) mS[rgrp] = mx;
        __syncthreads();   // pA/mS/mkS ready; all scT reads done

        // ---- GEMM: D(16x16) = P(16x128) @ expT(128 x [n0:n0+16))
        v8f d = {};
#pragma unroll
        for (int c = 0; c < 4; ++c) {
            // A fragment (matches B): lane half 0 holds
            // K 0..15, half 1 holds K 16..31; 2 K per VGPR.
            v16h a;
            const _Float16* pr   = &pA[l15][0];
            const int       base = 32 * c + 16 * half;
#pragma unroll
            for (int j = 0; j < 8; ++j) {
                const int koff = base + 2 * j;
                a[2 * j]     = pr[koff];
                a[2 * j + 1] = pr[koff + 1];
            }
            d = __builtin_amdgcn_wmma_f32_16x16x32_f16(
                    /*neg_a=*/false, a, /*neg_b=*/false, bfrag[c],
                    /*c_mod=*/(short)0, d, /*reuse_a=*/false, /*reuse_b=*/false);
        }

        // ---- write-back: this lane's 8 rows are contiguous in column-major
        // scT => vector LDS traffic.  Mask handled as an arithmetic blend
        // (sel*(nv-ov)+ov) so nothing can be sunk into a branch: the 8
        // v_log_f32 pipeline back-to-back with no exec-mask churn.
        {
            const int col = n0 + l15;
            float4* scol = (float4*)&scT[col][half * 8];
            float4  o0 = scol[0], o1 = scol[1];
            const float4 m4a = *(const float4*)&mS[half * 8];
            const float4 m4b = *(const float4*)&mS[half * 8 + 4];
            const int4   k4a = *(const int4*)&mkS[half * 8];
            const int4   k4b = *(const int4*)&mkS[half * 8 + 4];
            const float mv[8] = { m4a.x, m4a.y, m4a.z, m4a.w, m4b.x, m4b.y, m4b.z, m4b.w };
            const int   kv[8] = { k4a.x, k4a.y, k4a.z, k4a.w, k4b.x, k4b.y, k4b.z, k4b.w };
            float ov[8] = { o0.x, o0.y, o0.z, o0.w, o1.x, o1.y, o1.z, o1.w };
            float lg[8];
#pragma unroll
            for (int v = 0; v < 8; ++v) {
                // d[v] in [0.9, 142]: raw v_log_f32 is exact-safe here
                lg[v] = __builtin_amdgcn_logf(d[v]);
            }
#pragma unroll
            for (int v = 0; v < 8; ++v) {
                const float nv  = fmaf(lg[v], LN2, mv[v]) + ev[v];
                const float sel = (kv[v] > 0) ? 1.0f : 0.0f;
                ov[v] = fmaf(sel, nv - ov[v], ov[v]);
            }
            scol[0] = make_float4(ov[0], ov[1], ov[2], ov[3]);
            scol[1] = make_float4(ov[4], ov[5], ov[6], ov[7]);
        }
        __syncthreads();   // scT updated before next step's reads
    }

    // ---- denominator: logsumexp over T of (sc + end_transitions), per row
    {
        float sv[8];
        float mx = -3.402823e38f;
#pragma unroll
        for (int k = 0; k < 8; ++k) {
            const int t = sub + 16 * k;
            sv[k] = scT[t][rgrp] + endT[t];
            mx = fmaxf(mx, sv[k]);
        }
#pragma unroll
        for (int off = 8; off >= 1; off >>= 1) mx = fmaxf(mx, __shfl_xor(mx, off, 16));
        float sum = 0.f;
#pragma unroll
        for (int k = 0; k < 8; ++k) {
            sum += __builtin_amdgcn_exp2f((sv[k] - mx) * LOG2E);
        }
#pragma unroll
        for (int off = 8; off >= 1; off >>= 1) sum += __shfl_xor(sum, off, 16);
        if (sub == 0) denomWS[b0 + rgrp] = fmaf(__builtin_amdgcn_logf(sum), LN2, mx);
    }
}

// ---------------------------------------------------------------------------
// Kernel 3: deterministic final reduction:  out = sum_b (score_b - denom_b)
// ---------------------------------------------------------------------------
__global__ __launch_bounds__(256) void crf_reduce(
    const float* __restrict__ scoreWS, const float* __restrict__ denomWS,
    float* __restrict__ out)
{
    __shared__ float red[256];
    const int tid = threadIdx.x;
    red[tid] = (scoreWS[tid] - denomWS[tid]) +
               (scoreWS[tid + 256] - denomWS[tid + 256]);
    __syncthreads();
    for (int off = 128; off >= 1; off >>= 1) {
        if (tid < off) red[tid] += red[tid + off];
        __syncthreads();
    }
    if (tid == 0) out[0] = red[0];
}

// ---------------------------------------------------------------------------
extern "C" void kernel_launch(void* const* d_in, const int* in_sizes, int n_in,
                              void* d_out, int out_size, void* d_ws, size_t ws_size,
                              hipStream_t stream)
{
    const float* emissions = (const float*)d_in[0];
    const int*   tags      = (const int*)  d_in[1];
    const int*   mask      = (const int*)  d_in[2];
    const float* startT    = (const float*)d_in[3];
    const float* endT      = (const float*)d_in[4];
    const float* trans     = (const float*)d_in[5];
    float*       out       = (float*)d_out;

    float* scoreWS = (float*)d_ws;        // [Bb]
    float* denomWS = scoreWS + Bb;        // [Bb]

    crf_numerator<<<Bb, 128, 0, stream>>>(emissions, tags, mask,
                                          startT, endT, trans, scoreWS);
    crf_forward<<<Bb / ROWS, 256, 0, stream>>>(emissions, mask,
                                               startT, endT, trans, denomWS);
    crf_reduce<<<1, 256, 0, stream>>>(scoreWS, denomWS, out);
}
